// GCN_5016521802361
// MI455X (gfx1250) — compile-verified, matrix operand order checked by reading, not run on hardware
//
#include <hip/hip_runtime.h>
#include <hip/hip_bf16.h>

// ---------------------------------------------------------------------------
// Types for WMMA fragments (CDNA5 / gfx1250, wave32)
// ---------------------------------------------------------------------------
typedef _Float16      v16h  __attribute__((ext_vector_type(16)));
typedef float         v8f   __attribute__((ext_vector_type(8)));
typedef unsigned int  u32x4 __attribute__((ext_vector_type(4)));
typedef int           i32x4 __attribute__((ext_vector_type(4)));
typedef int           i32x8 __attribute__((ext_vector_type(8)));

#define GN   50000   // nodes
#define GD   16      // neighbors per node (= LSTM timesteps)
#define GF   128     // feature / hidden dim
#define GC   40      // classes
#define WP   136     // LDS row pitch in f16 (128 + 8 pad -> 4-bank advance/row)

#define HAS_ASYNC_LDS __has_builtin(__builtin_amdgcn_global_load_async_to_lds_b128)
#define HAS_TDM       __has_builtin(__builtin_amdgcn_tensor_load_to_lds)

// ---------------------------------------------------------------------------
// Fragment helpers.
// 16-bit A/B fragment layout (ISA 7.12.2): lane L -> row (L&15); half=(L>>4);
// VGPR0..3 hold K = 8*half + 0..7, VGPR4..7 hold K = 8*half + 16..23.
// => per lane: two contiguous 16-byte chunks at +0 and +16 elements.
// ---------------------------------------------------------------------------
__device__ __forceinline__ v16h ldfrag(const _Float16* p) {
    union { v16h h; u32x4 q[2]; } u;
    u.q[0] = *(const u32x4*)(p);
    u.q[1] = *(const u32x4*)(p + 16);
    return u.h;
}

__device__ __forceinline__ v8f wmma_f16(v16h a, v16h b, v8f c) {
    return __builtin_amdgcn_wmma_f32_16x16x32_f16(
        false, a, false, b, (short)0, c, false, false);
}

__device__ __forceinline__ v8f splat8(float x) {
    v8f r;
#pragma unroll
    for (int i = 0; i < 8; ++i) r[i] = x;
    return r;
}

__device__ __forceinline__ float sigm(float x)  { return 1.0f / (1.0f + __expf(-x)); }
__device__ __forceinline__ float tanhx(float x) { return 2.0f / (1.0f + __expf(-2.0f * x)) - 1.0f; }

// LDS byte offset of a __shared__ object (generic -> AS3 -> int)
__device__ __forceinline__ unsigned lds_addr_of(const void* p) {
    return (unsigned)(unsigned long long)(__attribute__((address_space(3))) const void*)p;
}

// ---- async global->LDS 16B copy (ASYNCcnt path), with plain fallback ------
// Builtin signature (from hipcc diagnostic): param0 is `int4 __device__ *`,
// so pass generic int4* pointers and let HIP's implicit AS conversion apply.
__device__ __forceinline__ void copy16_async(const _Float16* g, _Float16* l) {
#if HAS_ASYNC_LDS
    __builtin_amdgcn_global_load_async_to_lds_b128(
        (i32x4*)g, (i32x4*)l, 0, 0);
#else
    *(u32x4*)l = *(const u32x4*)g;
#endif
}

__device__ __forceinline__ void wait_async_all() {
#if HAS_ASYNC_LDS
#if __has_builtin(__builtin_amdgcn_s_wait_asynccnt)
    __builtin_amdgcn_s_wait_asynccnt(0);
#else
    asm volatile("s_wait_asynccnt 0x0" ::: "memory");
#endif
#endif
}

// ---- TDM: load one [512 x 128] f16 weight matrix into padded LDS ----------
// 2D D#: tile == tensor, data_size=2B, pad 4 DWORDs (16B) every 64 DWORDs
// (256B = one 128-f16 row)  -> LDS row pitch = 272B = WP f16.  (ISA ch.8 D#)
__device__ __forceinline__ void tdm_load_w(const _Float16* g, const _Float16* lds_dst) {
#if HAS_TDM
    unsigned long long ga = (unsigned long long)(const void*)g;
    u32x4 g0;
    g0[0] = 1u;                                   // count=1, user descriptor
    g0[1] = lds_addr_of(lds_dst);                 // lds_addr
    g0[2] = (unsigned)(ga & 0xFFFFFFFFull);       // global_addr[31:0]
    g0[3] = (unsigned)((ga >> 32) & 0x1FFFFFFull) // global_addr[56:32]
          | (2u << 30);                           // type = 2 ("image")
    i32x8 g1;
    g1[0] = (1 << 16)      // data_size = 2 bytes
          | (1 << 20)      // pad_enable
          | (5 << 22)      // pad_interval: 64 DWORDs
          | (3 << 25);     // pad_amount:   4 DWORDs
    g1[1] = (int)(128u << 16);   // tensor_dim0 = 128
    g1[2] = (int)(512u << 16);   // tensor_dim1 = 512
    g1[3] = (int)(128u << 16);   // tile_dim0   = 128
    g1[4] = 512;                 // tile_dim1   = 512 (tile_dim2 = 0)
    g1[5] = 128;                 // tensor_dim0_stride = 128
    g1[6] = 0;
    g1[7] = 0;
    i32x4 z4 = {0, 0, 0, 0};
#if __has_include(<hip/amd_detail/amd_gfx1250_TDM.h>)
    i32x8 z8 = {0, 0, 0, 0, 0, 0, 0, 0};
    __builtin_amdgcn_tensor_load_to_lds(g0, g1, z4, z4, z8, 0);
#else
    __builtin_amdgcn_tensor_load_to_lds(g0, g1, z4, z4, 0);
#endif
#else
    (void)g; (void)lds_dst;
#endif
}

__device__ __forceinline__ void wait_tensor_all() {
#if HAS_TDM
#if __has_builtin(__builtin_amdgcn_s_wait_tensorcnt)
    __builtin_amdgcn_s_wait_tensorcnt((short)0);
#else
    asm volatile("s_wait_tensorcnt 0x0" ::: "memory");
#endif
#endif
}

// ---------------------------------------------------------------------------
// f32 -> f16 conversion (weights + activations)
// ---------------------------------------------------------------------------
__global__ void cvt_f32_f16_kernel(const float* __restrict__ in,
                                   _Float16* __restrict__ out, int n) {
    int i = blockIdx.x * blockDim.x + threadIdx.x;
    if (i < n) out[i] = (_Float16)in[i];
}

// ---------------------------------------------------------------------------
// xp = relu(x @ Wp^T + bp)   [N,128] -> [N,128]  (f16 in, f16 out)
// 8 waves; wave w -> output cols [16w,16w+16).  Two accumulators (kt parity)
// so no WMMA->WMMA same-dest dependency.
// ---------------------------------------------------------------------------
__global__ __launch_bounds__(256, 1)
void proj_relu_kernel(const _Float16* __restrict__ xin,
                      const _Float16* __restrict__ wp,
                      const float*    __restrict__ bp,
                      _Float16*       __restrict__ xp) {
    const int tid  = threadIdx.x;
    const int wave = tid >> 5;
    const int lane = tid & 31;
    const int n    = lane & 15;
    const int hl   = lane >> 4;
    const size_t node = (size_t)blockIdx.x * 16;

    v8f a0 = splat8(bp[wave * 16 + n]);
    v8f a1 = splat8(0.0f);
#pragma unroll
    for (int kt = 0; kt < 4; ++kt) {
        v16h a = ldfrag(xin + (node + (size_t)n) * GF + kt * 32 + hl * 8);
        v16h b = ldfrag(wp  + (size_t)(wave * 16 + n) * GF + kt * 32 + hl * 8);
        if (kt & 1) a1 = wmma_f16(a, b, a1);
        else        a0 = wmma_f16(a, b, a0);
    }
    v8f acc = a0 + a1;
#pragma unroll
    for (int v = 0; v < 8; ++v) {
        int m = v + hl * 8;
        float val = acc[v];
        val = val > 0.0f ? val : 0.0f;
        xp[(node + (size_t)m) * GF + wave * 16 + n] = (_Float16)val;
    }
}

// ---------------------------------------------------------------------------
// Fused SAGE-LSTM tile kernel (16 nodes / workgroup, 8 waves).
// Wave w owns hidden cols [16w,16w+16) of all 4 gates -> register-local cell
// update.  Wih/Whh staged to LDS via TDM; neighbor gather via async LDS loads.
// 8 gate accumulators (Wih/Whh split) -> no WMMA same-dest back-to-back.
// ---------------------------------------------------------------------------
template <int OC, bool RELU_OUT, bool OUT_F16>
__global__ __launch_bounds__(256, 1)
void sage_lstm_kernel(const _Float16* __restrict__ xp,
                      const _Float16* __restrict__ xin,
                      const int*      __restrict__ edge_src,
                      const _Float16* __restrict__ wih,
                      const _Float16* __restrict__ whh,
                      const float*    __restrict__ bih,
                      const float*    __restrict__ bhh,
                      const _Float16* __restrict__ wl,
                      const float*    __restrict__ bl,
                      const _Float16* __restrict__ wr,
                      float*          __restrict__ out_f32,
                      _Float16*       __restrict__ out_f16)
{
    __shared__ alignas(16) _Float16 s_wih[512 * WP];   // 136 KiB
    __shared__ alignas(16) _Float16 s_whh[512 * WP];   // 136 KiB
    __shared__ alignas(16) _Float16 s_xt [16 * WP];
    __shared__ alignas(16) _Float16 s_h  [16 * WP];

    const int tid  = threadIdx.x;
    const int wave = tid >> 5;
    const int lane = tid & 31;
    const int n    = lane & 15;
    const int hl   = lane >> 4;
    const size_t node = (size_t)blockIdx.x * 16;

    // --- stage weights into LDS ---
#if HAS_TDM
    if (wave == 0) tdm_load_w(wih, s_wih);
    if (wave == 1) tdm_load_w(whh, s_whh);
#else
    for (int idx = tid; idx < 512 * 16; idx += 256) {
        int row = idx >> 4, c = idx & 15;
        *(u32x4*)&s_wih[row * WP + c * 8] = *(const u32x4*)&wih[(size_t)row * GF + c * 8];
        *(u32x4*)&s_whh[row * WP + c * 8] = *(const u32x4*)&whh[(size_t)row * GF + c * 8];
    }
#endif
    // --- zero h ---
    for (int i = tid; i < (16 * WP) / 8; i += 256) {
        u32x4 z = {0u, 0u, 0u, 0u};
        *(u32x4*)&s_h[i * 8] = z;
    }
    wait_tensor_all();
    __syncthreads();

    // gate biases (torch order i,f,g,o), broadcast over the node (m) axis
    float bias_g[4];
#pragma unroll
    for (int gt = 0; gt < 4; ++gt)
        bias_g[gt] = bih[gt * GF + wave * 16 + n] + bhh[gt * GF + wave * 16 + n];

    v8f c_state = splat8(0.0f);

    for (int t = 0; t < GD; ++t) {
        // --- gather timestep-t neighbor features into LDS (async path) ---
        {
            int row = tid >> 4, c = tid & 15;
            int src = edge_src[(node + (size_t)row) * GD + t];
            copy16_async(&xp[(size_t)src * GF + c * 8], &s_xt[row * WP + c * 8]);
        }
        wait_async_all();
        __syncthreads();

        // --- G = Xt @ Wih^T + H @ Whh^T + b : 8 independent accumulators ---
        v8f accW[4], accH[4];
#pragma unroll
        for (int gt = 0; gt < 4; ++gt) {
            accW[gt] = splat8(bias_g[gt]);
            accH[gt] = splat8(0.0f);
        }
#pragma unroll
        for (int kt = 0; kt < 4; ++kt) {
            v16h ax = ldfrag(&s_xt[n * WP + kt * 32 + hl * 8]);
            v16h ah = ldfrag(&s_h [n * WP + kt * 32 + hl * 8]);
#pragma unroll
            for (int gt = 0; gt < 4; ++gt) {
                int brow = gt * GF + wave * 16 + n;
                v16h bx = ldfrag(&s_wih[brow * WP + kt * 32 + hl * 8]);
                accW[gt] = wmma_f16(ax, bx, accW[gt]);
            }
#pragma unroll
            for (int gt = 0; gt < 4; ++gt) {
                int brow = gt * GF + wave * 16 + n;
                v16h bh = ldfrag(&s_whh[brow * WP + kt * 32 + hl * 8]);
                accH[gt] = wmma_f16(ah, bh, accH[gt]);
            }
        }
        __syncthreads();   // all reads of s_h / s_xt done before rewrite

        // --- cell update (register-local) + publish new h ---
#pragma unroll
        for (int v = 0; v < 8; ++v) {
            float ig = sigm(accW[0][v] + accH[0][v]);
            float fg = sigm(accW[1][v] + accH[1][v]);
            float gg = tanhx(accW[2][v] + accH[2][v]);
            float og = sigm(accW[3][v] + accH[3][v]);
            float cn = fg * c_state[v] + ig * gg;
            c_state[v] = cn;
            float hn = og * tanhx(cn);
            s_h[(v + 8 * hl) * WP + wave * 16 + n] = (_Float16)hn;
        }
        __syncthreads();
    }

    // --- epilogue: out = aggr @ Wl^T + bl + xin @ Wr^T (4 accumulators) ---
    constexpr int NCT = (OC + 15) / 16;
    if (wave < NCT) {
        int colbase = wave * 16;
        int bcol    = colbase + n < OC ? colbase + n : OC - 1;
        v8f racc[4];
        racc[0] = splat8(bl[bcol]);
        racc[1] = splat8(0.0f);
        racc[2] = splat8(0.0f);
        racc[3] = splat8(0.0f);
#pragma unroll
        for (int kt = 0; kt < 4; ++kt) {
            int base = (kt & 1) * 2;
            v16h ah = ldfrag(&s_h[n * WP + kt * 32 + hl * 8]);
            v16h ax = ldfrag(xin + (node + (size_t)n) * GF + kt * 32 + hl * 8);
            v16h bwl = ldfrag(wl + (size_t)bcol * GF + kt * 32 + hl * 8);
            racc[base] = wmma_f16(ah, bwl, racc[base]);
            v16h bwr = ldfrag(wr + (size_t)bcol * GF + kt * 32 + hl * 8);
            racc[base + 1] = wmma_f16(ax, bwr, racc[base + 1]);
        }
        v8f r = (racc[0] + racc[1]) + (racc[2] + racc[3]);
#pragma unroll
        for (int v = 0; v < 8; ++v) {
            int m   = v + 8 * hl;
            int col = colbase + n;
            float val = r[v];
            if (RELU_OUT) val = val > 0.0f ? val : 0.0f;
            if constexpr (OUT_F16) {
                out_f16[(node + (size_t)m) * GF + col] = (_Float16)val;
            } else {
                if (col < OC) out_f32[(node + (size_t)m) * OC + col] = val;
            }
        }
    }
}

// ---------------------------------------------------------------------------
// Host launcher
// ---------------------------------------------------------------------------
extern "C" void kernel_launch(void* const* d_in, const int* in_sizes, int n_in,
                              void* d_out, int out_size, void* d_ws, size_t ws_size,
                              hipStream_t stream) {
    (void)in_sizes; (void)n_in; (void)out_size; (void)ws_size;

    // 0:x 1:edge_src 2..7: dead MLP (unused)
    // 8:Wp1 9:bp1 10:Wih1 11:Whh1 12:bih1 13:bhh1 14:Wl1 15:bl1 16:Wr1
    // 17:Wp2 18:bp2 19:Wih2 20:Whh2 21:bih2 22:bhh2 23:Wl2 24:bl2 25:Wr2
    const int* esrc = (const int*)d_in[1];

    size_t off = 0;
    auto alloc = [&](size_t bytes) -> void* {
        off = (off + 255) & ~(size_t)255;
        void* p = (char*)d_ws + off;
        off += bytes;
        return p;
    };
    _Float16* x_f16  = (_Float16*)alloc((size_t)GN * GF * 2);
    _Float16* xp_f16 = (_Float16*)alloc((size_t)GN * GF * 2);
    _Float16* h1_f16 = (_Float16*)alloc((size_t)GN * GF * 2);
    _Float16* wp1f  = (_Float16*)alloc(GF * GF * 2);
    _Float16* wih1f = (_Float16*)alloc(4 * GF * GF * 2);
    _Float16* whh1f = (_Float16*)alloc(4 * GF * GF * 2);
    _Float16* wl1f  = (_Float16*)alloc(GF * GF * 2);
    _Float16* wr1f  = (_Float16*)alloc(GF * GF * 2);
    _Float16* wp2f  = (_Float16*)alloc(GF * GF * 2);
    _Float16* wih2f = (_Float16*)alloc(4 * GF * GF * 2);
    _Float16* whh2f = (_Float16*)alloc(4 * GF * GF * 2);
    _Float16* wl2f  = (_Float16*)alloc(GC * GF * 2);
    _Float16* wr2f  = (_Float16*)alloc(GC * GF * 2);

    auto cvt = [&](const void* src, _Float16* dst, int n) {
        cvt_f32_f16_kernel<<<(n + 255) / 256, 256, 0, stream>>>((const float*)src, dst, n);
    };
    cvt(d_in[0],  x_f16,  GN * GF);
    cvt(d_in[8],  wp1f,   GF * GF);
    cvt(d_in[10], wih1f,  4 * GF * GF);
    cvt(d_in[11], whh1f,  4 * GF * GF);
    cvt(d_in[14], wl1f,   GF * GF);
    cvt(d_in[16], wr1f,   GF * GF);
    cvt(d_in[17], wp2f,   GF * GF);
    cvt(d_in[19], wih2f,  4 * GF * GF);
    cvt(d_in[20], whh2f,  4 * GF * GF);
    cvt(d_in[23], wl2f,   GC * GF);
    cvt(d_in[25], wr2f,   GC * GF);

    const int tiles = GN / 16;  // 3125

    // ---- layer 1 ----
    proj_relu_kernel<<<tiles, 256, 0, stream>>>(x_f16, wp1f, (const float*)d_in[9], xp_f16);
    sage_lstm_kernel<GF, true, true><<<tiles, 256, 0, stream>>>(
        xp_f16, x_f16, esrc, wih1f, whh1f,
        (const float*)d_in[12], (const float*)d_in[13],
        wl1f, (const float*)d_in[15], wr1f,
        nullptr, h1_f16);

    // ---- layer 2 ----
    proj_relu_kernel<<<tiles, 256, 0, stream>>>(h1_f16, wp2f, (const float*)d_in[18], xp_f16);
    sage_lstm_kernel<GC, false, false><<<tiles, 256, 0, stream>>>(
        xp_f16, h1_f16, esrc, wih2f, whh2f,
        (const float*)d_in[21], (const float*)d_in[22],
        wl2f, (const float*)d_in[24], wr2f,
        (float*)d_out, nullptr);
}
